// RBM_37615323578531
// MI455X (gfx1250) — matrix operand order repaired and checked by reading, not compiled, not run on hardware
//
#include <hip/hip_runtime.h>
#include <hip/hip_bf16.h>

typedef __attribute__((ext_vector_type(16))) _Float16 v16h;
typedef __attribute__((ext_vector_type(8)))  _Float16 v8h;
typedef __attribute__((ext_vector_type(8)))  float    v8f;

#define RBM_B 8192
#define RBM_V 8192
#define RBM_H 256

__device__ __forceinline__ float rng_uniform(unsigned seed, unsigned idx) {
    unsigned x = idx ^ (seed * 0x9E3779B9u);
    x *= 0x7FEB352Du; x ^= x >> 15; x *= 0x846CA68Bu; x ^= x >> 16;
    return (float)(x >> 8) * (1.0f / 16777216.0f);
}

__device__ __forceinline__ float sigmoidf(float x) {
    return 1.0f / (1.0f + __expf(-x));
}

__device__ __forceinline__ v16h load_frag16(const _Float16* p, int gap) {
    // 16 halfs from two 16B chunks at p and p+gap (gap in halfs)
    v8h lo = *(const v8h*)(p);
    v8h hi = *(const v8h*)(p + gap);
    return __builtin_shufflevector(lo, hi, 0, 1, 2, 3, 4, 5, 6, 7,
                                   8, 9, 10, 11, 12, 13, 14, 15);
}

// ---------------------------------------------------------------------------
// f32 -> f16 conversion, 8 elements/thread (32B in, 16B out)
// ---------------------------------------------------------------------------
__global__ void __launch_bounds__(256)
cvt_f32_f16_kernel(const float* __restrict__ in, _Float16* __restrict__ out, size_t n8) {
    size_t i = ((size_t)blockIdx.x * 256 + threadIdx.x);
    if (i >= n8) return;
    const float4 v0 = ((const float4*)in)[i * 2];
    const float4 v1 = ((const float4*)in)[i * 2 + 1];
    v8h o;
    o[0] = (_Float16)v0.x; o[1] = (_Float16)v0.y; o[2] = (_Float16)v0.z; o[3] = (_Float16)v0.w;
    o[4] = (_Float16)v1.x; o[5] = (_Float16)v1.y; o[6] = (_Float16)v1.z; o[7] = (_Float16)v1.w;
    ((v8h*)out)[i] = o;
}

// ---------------------------------------------------------------------------
// w (V x H, f32) -> wh (V x H, f16) and wT (H x V, f16), tiled through LDS
// ---------------------------------------------------------------------------
__global__ void __launch_bounds__(256)
wprep_kernel(const float* __restrict__ w, _Float16* __restrict__ wh, _Float16* __restrict__ wT) {
    __shared__ _Float16 tile[32][33];
    const int tx = threadIdx.x & 31;
    const int ty = threadIdx.x >> 5;
    const int v0 = blockIdx.x * 32;    // along V
    const int h0 = blockIdx.y * 32;    // along H
#pragma unroll
    for (int j = 0; j < 4; ++j) {
        int vv = v0 + ty + j * 8;
        float x = w[(size_t)vv * RBM_H + h0 + tx];
        _Float16 hx = (_Float16)x;
        tile[ty + j * 8][tx] = hx;
        wh[(size_t)vv * RBM_H + h0 + tx] = hx;
    }
    __syncthreads();
#pragma unroll
    for (int j = 0; j < 4; ++j) {
        int hh = h0 + ty + j * 8;
        wT[(size_t)hh * RBM_V + v0 + tx] = tile[tx][ty + j * 8];
    }
}

// ---------------------------------------------------------------------------
// D = act(bias + A * Bsrc^T): A (M x K) f16 row-major, Bsrc (N x K) f16
// row-major. Software-pipelined with role-alternating fragment buffers
// (2x K-slice unroll) -> no register copies between slices, partial
// s_wait_loadcnt before each WMMA group. K/32 must be even (K=256 or 8192).
// mode 0: out = sigmoid(..) f16 probs; mode 1: out = bernoulli f16 {0,1}.
// Each wave computes a 16x64 strip (4 WMMA N-tiles per A fragment).
// ---------------------------------------------------------------------------
#define GNT 4
__global__ void __launch_bounds__(256)
gemm_abT_kernel(const _Float16* __restrict__ A, const _Float16* __restrict__ Bs,
                const float* __restrict__ bias, _Float16* __restrict__ out,
                int M, int N, int K, int mode, unsigned seed) {
    const int lane = threadIdx.x & 31;
    const int wave = threadIdx.x >> 5;
    const int nGroups = N / (16 * GNT);
    const long task = (long)blockIdx.x * 8 + wave;
    const int mT = (int)(task / nGroups);
    const int ng = (int)(task % nGroups);
    const int mBase = mT * 16;
    const int nBase = ng * 16 * GNT;

    const int l15 = lane & 15;
    const int aK  = (lane < 16) ? 0 : 8;    // 16-bit A fragment K offset
    const int bK  = (lane < 16) ? 0 : 8;    // 16-bit B fragment K offset

    v8f acc[GNT];
    const v8f zero = {0.f, 0.f, 0.f, 0.f, 0.f, 0.f, 0.f, 0.f};
#pragma unroll
    for (int t = 0; t < GNT; ++t) acc[t] = zero;

    const _Float16* pA  = A + (size_t)(mBase + l15) * K + aK;
    const _Float16* pB0 = Bs + (size_t)(nBase + l15) * K + bK;

    v16h af0, af1;
    v16h bf0[GNT], bf1[GNT];

    auto load_slice = [&](int k, v16h& af, v16h (&bf)[GNT]) {
        af = load_frag16(pA + k, 16);
#pragma unroll
        for (int t = 0; t < GNT; ++t)
            bf[t] = load_frag16(pB0 + (size_t)(t * 16) * K + k, 16);
    };
    auto compute = [&](const v16h& af, const v16h (&bf)[GNT]) {
#pragma unroll
        for (int t = 0; t < GNT; ++t)
            acc[t] = __builtin_amdgcn_wmma_f32_16x16x32_f16(
                false, af, false, bf[t], (short)0, acc[t], false, false);
    };

    load_slice(0, af0, bf0);
    int k = 32;
    for (; k + 32 < K; k += 64) {
        load_slice(k, af1, bf1);
        __builtin_prefetch(pA + k + 256, 0, 1);
        compute(af0, bf0);
        load_slice(k + 32, af0, bf0);
        compute(af1, bf1);
    }
    // tail: k == K - 32 (slice count even)
    load_slice(k, af1, bf1);
    compute(af0, bf0);
    compute(af1, bf1);

    const int rowTop = (lane < 16) ? 0 : 8;
#pragma unroll
    for (int t = 0; t < GNT; ++t) {
        const int nn = nBase + t * 16 + l15;
        const float bv = bias[nn];
#pragma unroll
        for (int r = 0; r < 8; ++r) {
            const int mm = mBase + rowTop + r;
            const float p = sigmoidf(acc[t][r] + bv);
            _Float16 val;
            if (mode == 0) {
                val = (_Float16)p;
            } else {
                float u = rng_uniform(seed, (unsigned)((size_t)mm * (unsigned)N + (unsigned)nn));
                val = (u < p) ? (_Float16)1.0f : (_Float16)0.0f;
            }
            out[(size_t)mm * N + nn] = val;
        }
    }
}

// ---------------------------------------------------------------------------
// D = scale * X^T * Y : X (Kd x M) f16, Y (Kd x N) f16, D (M x N) f32.
// Double-buffered LDS staging (one barrier per K-slice); staging uses
// b128 global loads + b128 LDS stores (row strides padded to 16B multiples).
// Block: 8 waves -> 64(M) x 32(N) tile.
// ---------------------------------------------------------------------------
#define ATM 64
#define ATN 32
#define XPAD 8   // row stride 72 halfs = 144B (16B multiple)
#define YPAD 8   // row stride 40 halfs = 80B  (16B multiple)
__global__ void __launch_bounds__(256)
gemm_AT_kernel(const _Float16* __restrict__ X, const _Float16* __restrict__ Y,
               float* __restrict__ D, int Kd, int M, int N, float scale) {
    __shared__ _Float16 Xs[2][32][ATM + XPAD];
    __shared__ _Float16 Ys[2][32][ATN + YPAD];
    const int tid  = threadIdx.x;
    const int lane = tid & 31;
    const int wave = tid >> 5;
    const int wm = wave & 3;   // 4 waves along M
    const int wn = wave >> 2;  // 2 waves along N
    const int m0 = blockIdx.x * ATM;
    const int n0 = blockIdx.y * ATN;
    const int l15 = lane & 15;
    const int aK = (lane < 16) ? 0 : 8;
    const int bK = (lane < 16) ? 0 : 8;

    // staging coordinates: X tile 32x64 halfs -> 256 threads x v8h
    const int xr = tid >> 3, xc = (tid & 7) * 8;
    // Y tile 32x32 halfs -> 128 threads x v8h
    const int yr = tid >> 2, yc = (tid & 3) * 8;

    auto stage = [&](int buf, int k0) {
        v8h xv = *(const v8h*)(X + (size_t)(k0 + xr) * M + m0 + xc);
        *(v8h*)(&Xs[buf][xr][xc]) = xv;
        if (tid < 128) {
            v8h yv = *(const v8h*)(Y + (size_t)(k0 + yr) * N + n0 + yc);
            *(v8h*)(&Ys[buf][yr][yc]) = yv;
        }
    };

    v8f acc = {0.f, 0.f, 0.f, 0.f, 0.f, 0.f, 0.f, 0.f};

    stage(0, 0);
    __syncthreads();
    int cur = 0;
    for (int k0 = 0; k0 < Kd; k0 += 32) {
        if (k0 + 32 < Kd) stage(1 - cur, k0 + 32);

        const int ml = wm * 16 + l15;
        const int nl = wn * 16 + l15;
        v16h af, bf;
#pragma unroll
        for (int i = 0; i < 8; ++i) {
            af[i]     = Xs[cur][aK + i][ml];
            af[i + 8] = Xs[cur][aK + 16 + i][ml];
        }
#pragma unroll
        for (int i = 0; i < 8; ++i) {
            bf[i]     = Ys[cur][bK + i][nl];
            bf[i + 8] = Ys[cur][bK + 16 + i][nl];
        }

        acc = __builtin_amdgcn_wmma_f32_16x16x32_f16(
            false, af, false, bf, (short)0, acc, false, false);
        __syncthreads();
        cur ^= 1;
    }

    const int rowTop = (lane < 16) ? 0 : 8;
    const int nn = n0 + wn * 16 + l15;
#pragma unroll
    for (int r = 0; r < 8; ++r) {
        const int mm = m0 + wm * 16 + rowTop + r;
        D[(size_t)mm * N + nn] = acc[r] * scale;
    }
}

// ---------------------------------------------------------------------------
// Column mean over rows: out[c] = scale * sum_r in[r*cols + c]
// ---------------------------------------------------------------------------
template <typename T>
__global__ void __launch_bounds__(256)
colmean_kernel(const T* __restrict__ in, float* __restrict__ out,
               int rows, int cols, float scale) {
    const int c = blockIdx.x * 256 + threadIdx.x;
    if (c >= cols) return;
    float s = 0.f;
    for (int r = 0; r < rows; ++r) s += (float)in[(size_t)r * cols + c];
    out[c] = s * scale;
}

// ---------------------------------------------------------------------------
// Bernoulli sampling of an f16 probability buffer -> f16 {0,1}
// ---------------------------------------------------------------------------
__global__ void __launch_bounds__(256)
bernoulli_kernel(const _Float16* __restrict__ p, _Float16* __restrict__ o,
                 unsigned n, unsigned seed) {
    unsigned i = blockIdx.x * 256 + threadIdx.x;
    if (i >= n) return;
    float u = rng_uniform(seed, i);
    o[i] = (u < (float)p[i]) ? (_Float16)1.0f : (_Float16)0.0f;
}

// ---------------------------------------------------------------------------
extern "C" void kernel_launch(void* const* d_in, const int* in_sizes, int n_in,
                              void* d_out, int out_size, void* d_ws, size_t ws_size,
                              hipStream_t stream) {
    const int B = RBM_B, V = RBM_V, H = RBM_H;
    const float* v    = (const float*)d_in[0];  // (B,V)
    const float* w    = (const float*)d_in[1];  // (V,H)
    const float* avec = (const float*)d_in[2];  // (V,)
    const float* bvec = (const float*)d_in[3];  // (H,)

    float* out      = (float*)d_out;
    float* vh_data  = out;                          // V*H
    float* vh_model = out + (size_t)V * H;          // V*H
    float* v_data   = out + (size_t)2 * V * H;      // V
    float* v_model  = v_data + V;                   // V
    float* h_data   = v_model + V;                  // H
    float* h_model  = h_data + H;                   // H

    // workspace carving
    size_t off = 0;
    char* ws = (char*)d_ws;
    auto carve = [&](size_t bytes) -> void* {
        void* p = ws + off;
        off += (bytes + 255) & ~(size_t)255;
        return p;
    };
    _Float16* vh  = (_Float16*)carve((size_t)B * V * 2);  // f16 copy of v
    _Float16* wh  = (_Float16*)carve((size_t)V * H * 2);  // f16 w (V,H)
    _Float16* wT  = (_Float16*)carve((size_t)V * H * 2);  // f16 w^T (H,V)
    _Float16* ph  = (_Float16*)carve((size_t)B * H * 2);  // p_h probabilities
    _Float16* hs0 = (_Float16*)carve((size_t)B * H * 2);  // sampled h ping
    _Float16* hs1 = (_Float16*)carve((size_t)B * H * 2);  // sampled h pong
    _Float16* vs  = (_Float16*)carve((size_t)B * V * 2);  // sampled v

    // 1) precision conversion / weight transpose
    {
        size_t n8 = ((size_t)B * V) / 8;
        cvt_f32_f16_kernel<<<(unsigned)(n8 / 256), 256, 0, stream>>>(v, vh, n8);
        dim3 g(V / 32, H / 32);
        wprep_kernel<<<g, 256, 0, stream>>>(w, wh, wT);
    }

    // 2) positive phase: p_h = sigmoid(b + v @ w)   [A=vh, Bsrc=wT]
    {
        long tasks = (long)(B / 16) * (H / 64);
        gemm_abT_kernel<<<(unsigned)(tasks / 8), 256, 0, stream>>>(
            vh, wT, bvec, ph, B, H, V, /*mode=*/0, 0u);
    }

    // 3) data statistics
    colmean_kernel<float><<<V / 256, 256, 0, stream>>>(v, v_data, B, V, 1.0f / B);
    colmean_kernel<_Float16><<<1, 256, 0, stream>>>(ph, h_data, B, H, 1.0f / B);

    // 4) initial h sample
    bernoulli_kernel<<<(B * H) / 256, 256, 0, stream>>>(ph, hs0, (unsigned)(B * H), 0x1234567u);

    // 5) CD-k Gibbs chain (k = 5), fused sigmoid+bernoulli epilogues
    _Float16* hcur = hs0;
    _Float16* hnxt = hs1;
    for (int step = 0; step < 5; ++step) {
        // v_s = bernoulli(sigmoid(a + h @ w^T))  [A=hcur (B,H), Bsrc=wh (V,H)]
        {
            long tasks = (long)(B / 16) * (V / 64);
            gemm_abT_kernel<<<(unsigned)(tasks / 8), 256, 0, stream>>>(
                hcur, wh, avec, vs, B, V, H, /*mode=*/1, 0xA0000000u + step);
        }
        // h_s = bernoulli(sigmoid(b + v_s @ w))  [A=vs (B,V), Bsrc=wT (H,V)]
        {
            long tasks = (long)(B / 16) * (H / 64);
            gemm_abT_kernel<<<(unsigned)(tasks / 8), 256, 0, stream>>>(
                vs, wT, bvec, hnxt, B, H, V, /*mode=*/1, 0xB0000000u + step);
        }
        _Float16* t = hcur; hcur = hnxt; hnxt = t;
    }

    // 6) model statistics
    colmean_kernel<_Float16><<<V / 256, 256, 0, stream>>>(vs, v_model, B, V, 1.0f / B);
    colmean_kernel<_Float16><<<1, 256, 0, stream>>>(hcur, h_model, B, H, 1.0f / B);

    // 7) correlation GEMMs (K = batch): vh_data = v^T @ p_h / B, vh_model = v_s^T @ h_s / B
    {
        dim3 g(V / ATM, H / ATN);
        gemm_AT_kernel<<<g, 256, 0, stream>>>(vh, ph, vh_data, B, V, H, 1.0f / B);
        gemm_AT_kernel<<<g, 256, 0, stream>>>(vs, hcur, vh_model, B, V, H, 1.0f / B);
    }

    (void)in_sizes; (void)n_in; (void)out_size; (void)ws_size;
}
